// ModalityMoERouter_78288663872365
// MI455X (gfx1250) — matrix operand order repaired and checked by reading, not compiled, not run on hardware
//
#include <hip/hip_runtime.h>
#include <hip/hip_bf16.h>

typedef __bf16 v16bf  __attribute__((ext_vector_type(16)));
typedef float  v8f    __attribute__((ext_vector_type(8)));
typedef __bf16 bf16x2 __attribute__((ext_vector_type(2)));
typedef unsigned int uint2v __attribute__((ext_vector_type(2)));
typedef unsigned int uint4v __attribute__((ext_vector_type(4)));
typedef int i32x4g __attribute__((vector_size(16)));   // GCC-style, matches builtin

#define BB 16
#define NN 8192
#define DD 512
#define HH 256
#define EE 16
#define KTOT 515
#define NCHUNK 17          // 17 * 32 = 544 >= 515
#define TOTAL_TOK (BB*NN)  // 131072
#define BNE (BB*NN*EE)     // 2097152
#define KSEL 1024

// W1f: NCHUNK chunks, each 512 frags ([tile(16)][lane(32)]) of 16 bf16 = 16KB/chunk
#define W1F_FRAGS (NCHUNK*512)
#define W2F_FRAGS 256

#if defined(__has_builtin)
#if __has_builtin(__builtin_amdgcn_global_load_async_to_lds_b128)
#define ASYNC_LDS 1
#endif
#endif

__device__ __forceinline__ void cp_b128(void* lds, const void* g) {
#ifdef ASYNC_LDS
    __builtin_amdgcn_global_load_async_to_lds_b128(
        (__attribute__((address_space(1))) i32x4g*)g,
        (__attribute__((address_space(3))) i32x4g*)lds, 0, 0);
#else
    *(uint4v*)lds = *(const uint4v*)g;
#endif
}
__device__ __forceinline__ void cp_wait() {
#ifdef ASYNC_LDS
#if __has_builtin(__builtin_amdgcn_s_wait_asynccnt)
    __builtin_amdgcn_s_wait_asynccnt(0);
#else
    asm volatile("s_wait_asynccnt 0" ::: "memory");
#endif
#endif
}

__device__ __forceinline__ unsigned pack2(float a, float b) {
    bf16x2 v = { (__bf16)a, (__bf16)b };
    return __builtin_bit_cast(unsigned, v);
}

__device__ __forceinline__ float gate_in(const float* __restrict__ tokens,
                                         const float* __restrict__ xyz,
                                         long tok, int k) {
    if (k < DD) return tokens[tok * DD + k];
    if (k < KTOT) return xyz[tok * 3 + (k - DD)];
    return 0.f;
}

// ---------------- Kernel 0: pre-convert W1/W2 to bf16 fragment layout -------
__global__ void k0_prep(const float* __restrict__ W1,
                        const float* __restrict__ W2,
                        __bf16* __restrict__ W1f,
                        __bf16* __restrict__ W2f) {
    const int gid = blockIdx.x * 256 + threadIdx.x;
    if (gid < W1F_FRAGS) {
        const int chunk = gid >> 9;
        const int f = gid & 511;
        const int tile = f >> 5, lane2 = f & 31;
        const int h2 = lane2 >> 4, col = tile * 16 + (lane2 & 15);
        v16bf w;
#pragma unroll
        for (int e = 0; e < 16; ++e) {
            const int k = chunk * 32 + (e >= 8 ? 16 : 0) + h2 * 8 + (e & 7);           // B frag: K = (e>=8?16:0)+half*8+(e&7)
            w[e] = (__bf16)((k < KTOT) ? W1[(long)k * HH + col] : 0.f);
        }
        ((v16bf*)W1f)[gid] = w;
    } else if (gid < W1F_FRAGS + W2F_FRAGS) {
        const int g2 = gid - W1F_FRAGS;
        const int k0g = g2 >> 5, lane2 = g2 & 31;
        const int h2 = lane2 >> 4, nn = lane2 & 15;
        v16bf w;
#pragma unroll
        for (int e = 0; e < 16; ++e)
            w[e] = (__bf16)W2[(k0g * 32 + (e >= 8 ? 16 : 0) + h2 * 8 + (e & 7)) * EE + nn];
        ((v16bf*)W2f)[g2] = w;
    }
}

// ---------------- Kernel 1: distances + global sum ----------------
__global__ void k1_dist(const float* __restrict__ xyz,
                        const float* __restrict__ centers,
                        float* __restrict__ dists,
                        float* __restrict__ dist_sum) {
    __shared__ float sc[EE * 3];
    __shared__ float red[256];
    const int tid = threadIdx.x;
    if (tid < EE * 3) sc[tid] = centers[tid];
    __syncthreads();
    const long tok = (long)blockIdx.x * 256 + tid;
    const float x = xyz[tok * 3 + 0];
    const float y = xyz[tok * 3 + 1];
    const float z = xyz[tok * 3 + 2];
    float local = 0.f;
#pragma unroll
    for (int e = 0; e < EE; ++e) {
        const float dx = x - sc[e * 3 + 0];
        const float dy = y - sc[e * 3 + 1];
        const float dz = z - sc[e * 3 + 2];
        const float d = sqrtf(dx * dx + dy * dy + dz * dz);
        dists[tok * EE + e] = d;
        local += d;
    }
    red[tid] = local;
    __syncthreads();
    for (int s = 128; s > 0; s >>= 1) {
        if (tid < s) red[tid] += red[tid + s];
        __syncthreads();
    }
    if (tid == 0) atomicAdd(dist_sum, red[0]);
}

// ---------------- Kernel 2: fused gate MLP (bf16 WMMA) ----------------------
// 512 threads = 16 waves, 64 tokens/block; waves = 4 row-tiles x 4 col-groups.
// B operand: async contiguous copy of pre-swizzled W1f chunk (16KB) into LDS,
// double buffered. A operand: staged+converted into fragment order (b64 store).
// h written in GEMM2 A-fragment order (aliases B buffers); waves 0..3 do GEMM2.
__global__ void __launch_bounds__(512)
k2_mlp(const float* __restrict__ tokens,
       const float* __restrict__ xyz,
       const __bf16* __restrict__ W1f,
       const float* __restrict__ b1,
       const __bf16* __restrict__ W2f,
       const float* __restrict__ b2,
       const float* __restrict__ dists,
       const float* __restrict__ dist_sum,
       float* __restrict__ logits) {
    __shared__ v16bf sBfrag[2][16 * 32];   // 32KB (aliased by sHfrag after GEMM1)
    __shared__ v16bf sAfrag[2][4 * 32];    // 8KB
    __shared__ v16bf sW2frag[8 * 32];      // 8KB

    const int tid  = threadIdx.x;
    const int wave = tid >> 5;
    const int lane = tid & 31;
    const int half = lane >> 4;
    const int ln   = lane & 15;
    const long tok0 = (long)blockIdx.x * 64;

    const int rowTile = wave & 3;
    const int colGrp  = wave >> 2;

    v8f acc[4];
#pragma unroll
    for (int t = 0; t < 4; ++t)
#pragma unroll
        for (int j = 0; j < 8; ++j) acc[t][j] = 0.f;

    auto stageB = [&](int buf, int chunk) {
        // contiguous 16KB copy: 512 threads x 32B
        const char* g = (const char*)W1f + (size_t)chunk * 16384 + tid * 32;
        char* l = (char*)&sBfrag[buf][0] + tid * 32;
        cp_b128(l, g);
        cp_b128(l + 16, g + 16);
    };
    auto stageA = [&](int buf, int k0) {
        // frag elem e of [rowTile][lane] = gate_in[tok, k0+(e>=8?16:0)+half*8+(e&7)]
        const int run  = tid >> 1;                 // 256 runs
        const int part = tid & 1;
        const int rT   = run >> 6;                 // 0..3
        const int lA   = (run >> 1) & 31;
        const int seg  = run & 1;                  // e-base 0 or 8
        const int hA   = lA >> 4;
        const long tok = tok0 + rT * 16 + (lA & 15);
        const int kb   = k0 + seg * 16 + hA * 8 + part * 4;
        float v[4];
#pragma unroll
        for (int i = 0; i < 4; ++i) v[i] = gate_in(tokens, xyz, tok, kb + i);
        uint2v p = { pack2(v[0], v[1]), pack2(v[2], v[3]) };
        __bf16* dst = (__bf16*)&sAfrag[buf][rT * 32 + lA];
        *(uint2v*)(dst + seg * 8 + part * 4) = p;
    };

    // W2 fragments: contiguous 8KB copy
    cp_b128((char*)sW2frag + tid * 16, (const char*)W2f + tid * 16);

    stageB(0, 0);
    stageA(0, 0);
    cp_wait();
    __syncthreads();
    for (int c = 0; c < NCHUNK; ++c) {
        const int buf = c & 1;
        if (c + 1 < NCHUNK) {
            stageB(buf ^ 1, c + 1);
            stageA(buf ^ 1, (c + 1) * 32);
        }
        const v16bf af = sAfrag[buf][rowTile * 32 + lane];
#pragma unroll
        for (int t = 0; t < 4; ++t) {
            const v16bf bf = sBfrag[buf][(colGrp * 4 + t) * 32 + lane];
            acc[t] = __builtin_amdgcn_wmma_f32_16x16x32_bf16(
                false, af, false, bf, (short)0, acc[t], false, false);
        }
        cp_wait();
        __syncthreads();
    }

    // ---- epilogue 1: bias + exact GELU -> h in GEMM2 A-fragment order ------
    v16bf* sHfrag = &sBfrag[0][0];   // 32KB alias: [(w2*8+k0g)*32 + l2]
#pragma unroll
    for (int t = 0; t < 4; ++t) {
#pragma unroll
        for (int v = 0; v < 8; ++v) {
            const int row = rowTile * 16 + half * 8 + v;          // C/D: M = v+half*8
            const int col = (colGrp * 4 + t) * 16 + ln;
            const float x = acc[t][v] + b1[col];
            const float g = 0.5f * x * (1.f + erff(x * 0.70710678118654752f));
            const int w2  = row >> 4, rl = row & 15;
            const int k0g = col >> 5, rem = col & 31;
            const int seg = rem >> 4, h2 = (rem >> 3) & 1;
            const int e   = seg * 8 + (rem & 7);
            ((__bf16*)&sHfrag[(w2 * 8 + k0g) * 32 + h2 * 16 + rl])[e] = (__bf16)g;
        }
    }
    __syncthreads();

    // ---- GEMM2: waves 0..3 compute one 16x16 logits tile each (K=256) ------
    if (wave < 4) {
        v8f c2;
#pragma unroll
        for (int j = 0; j < 8; ++j) c2[j] = 0.f;
#pragma unroll
        for (int k0g = 0; k0g < 8; ++k0g) {
            const v16bf af = sHfrag[(wave * 8 + k0g) * 32 + lane];
            const v16bf bf = sW2frag[k0g * 32 + lane];
            c2 = __builtin_amdgcn_wmma_f32_16x16x32_bf16(
                false, af, false, bf, (short)0, c2, false, false);
        }
        const float inv_scale = 1.f / (dist_sum[0] * (1.f / (float)BNE) + 1e-6f);
#pragma unroll
        for (int v = 0; v < 8; ++v) {
            const long tok = tok0 + wave * 16 + half * 8 + v;
            const float cl = c2[v] + b2[ln];
            logits[tok * EE + ln] = cl - dists[tok * EE + ln] * inv_scale;
        }
    }
}

// ---------------- Kernel 3: per-(b,e) radix select of 1024th-largest --------
__device__ __forceinline__ unsigned f2mono(float f) {
    const unsigned u = __float_as_uint(f);
    return (u & 0x80000000u) ? ~u : (u | 0x80000000u);
}

__global__ void k3_select(const float* __restrict__ logits,
                          unsigned* __restrict__ thresh) {
    const int be = blockIdx.x;             // b*16 + e
    const int b = be >> 4, e = be & 15;
    const float* base = logits + (long)b * NN * EE + e;
    __shared__ unsigned hist[256];
    __shared__ unsigned sh_prefix, sh_want;
    const int tid = threadIdx.x;
    if (tid == 0) { sh_prefix = 0u; sh_want = KSEL; }
    __syncthreads();
    for (int pass = 3; pass >= 0; --pass) {
        hist[tid] = 0u;
        __syncthreads();
        const unsigned prefix = sh_prefix;
        const unsigned hmask = (pass == 3) ? 0u : (0xFFFFFFFFu << (8 * (pass + 1)));
        for (int n = tid; n < NN; n += 256) {
            const unsigned key = f2mono(base[(long)n * EE]);
            if ((key & hmask) == (prefix & hmask)) {
                atomicAdd(&hist[(key >> (8 * pass)) & 255u], 1u);
            }
        }
        __syncthreads();
        if (tid == 0) {
            const unsigned want = sh_want;
            unsigned cum = 0; int bin = 0;
            for (int i = 255; i >= 0; --i) {
                const unsigned c = hist[i];
                if (cum + c >= want) { bin = i; sh_want = want - cum; break; }
                cum += c;
            }
            sh_prefix = prefix | ((unsigned)bin << (8 * pass));
        }
        __syncthreads();
    }
    if (tid == 0) thresh[be] = sh_prefix;   // select keys >= thresh
}

// ---------------- Kernel 4: gate + floor + cap redistribution + normalize ---
__global__ void k4_final(const float* __restrict__ logits,
                         const unsigned* __restrict__ thresh,
                         const int* __restrict__ tarr,
                         float* __restrict__ out) {
    const long tok = (long)blockIdx.x * 256 + threadIdx.x;
    const int b = (int)(tok >> 13);   // N = 8192
    __shared__ unsigned th[EE];
    __shared__ float s_cap;
    if (threadIdx.x < EE) th[threadIdx.x] = thresh[b * EE + threadIdx.x];
    if (threadIdx.x == 0) s_cap = 0.5f + 0.1f * (float)tarr[b] * (1.f / 1000.f);
    __syncthreads();
    const float cap = s_cap;

    float d[EE];
    float esum = 0.f, hsum = 0.f;
#pragma unroll
    for (int e = 0; e < EE; ++e) {
        const float x = logits[tok * EE + e];
        float disp = (f2mono(x) >= th[e]) ? (1.f / (1.f + expf(-x))) : 0.f;
        disp = 0.4f * disp + 0.0375f;            // alpha = 0.6
        const float ex = fmaxf(disp - cap, 0.f);
        const float cp = disp - ex;
        esum += ex;
        hsum += fmaxf(cap - cp, 0.f);
        d[e] = cp;
    }
    hsum = fmaxf(hsum, 1e-8f);
    float dsum = 0.f;
#pragma unroll
    for (int e = 0; e < EE; ++e) {
        d[e] = d[e] + esum * fmaxf(cap - d[e], 0.f) / hsum;
        dsum += d[e];
    }
    const float inv = 1.f / (dsum + 1e-8f);
    const long base = tok * EE;
#pragma unroll
    for (int e = 0; e < EE; ++e) {
        out[base + e] = d[e];                     // dispatch
        out[(long)BNE + base + e] = d[e] * inv;   // combine
    }
}

extern "C" void kernel_launch(void* const* d_in, const int* in_sizes, int n_in,
                              void* d_out, int out_size, void* d_ws, size_t ws_size,
                              hipStream_t stream) {
    (void)in_sizes; (void)n_in; (void)out_size; (void)ws_size;
    const float* tokens  = (const float*)d_in[0];
    const float* xyz     = (const float*)d_in[1];
    const int*   t       = (const int*)  d_in[2];
    const float* W1      = (const float*)d_in[3];
    const float* b1      = (const float*)d_in[4];
    const float* W2      = (const float*)d_in[5];
    const float* b2      = (const float*)d_in[6];
    const float* centers = (const float*)d_in[7];
    float* out = (float*)d_out;

    char* base = (char*)d_ws;
    float*    dists    = (float*)base;                                 // 8MB
    float*    logits   = (float*)(base + (size_t)BNE * 4);             // 8MB
    float*    dist_sum = (float*)(base + (size_t)2 * BNE * 4);         // 4B
    unsigned* thresh   = (unsigned*)(base + (size_t)2 * BNE * 4 + 64); // 1KB
    __bf16*   W1f      = (__bf16*)(base + (size_t)2 * BNE * 4 + 4096); // 272KB
    __bf16*   W2f      = (__bf16*)(base + (size_t)2 * BNE * 4 + 4096
                                   + (size_t)W1F_FRAGS * 32);          // 8KB

    (void)hipMemsetAsync(dist_sum, 0, sizeof(float), stream);

    k0_prep<<<(W1F_FRAGS + W2F_FRAGS) / 256, 256, 0, stream>>>(W1, W2, W1f, W2f);
    k1_dist<<<TOTAL_TOK / 256, 256, 0, stream>>>(xyz, centers, dists, dist_sum);
    k2_mlp<<<TOTAL_TOK / 64, 512, 0, stream>>>(tokens, xyz, W1f, b1, W2f, b2,
                                               dists, dist_sum, logits);
    k3_select<<<BB * EE, 256, 0, stream>>>(logits, thresh);
    k4_final<<<TOTAL_TOK / 256, 256, 0, stream>>>(logits, thresh, t, out);
}
